// BatchMatchedMSELoss_84086869721402
// MI455X (gfx1250) — matrix operand order repaired and checked by reading, not compiled, not run on hardware
//
#include <hip/hip_runtime.h>
#include <stdint.h>

// ---- types ----------------------------------------------------------------
typedef __attribute__((ext_vector_type(8)))  float        v8f;
typedef __attribute__((ext_vector_type(16))) __bf16       v16bf;
typedef __attribute__((ext_vector_type(4)))  float        f32x4;
typedef __attribute__((ext_vector_type(4)))  unsigned int u32x4;
typedef __attribute__((ext_vector_type(2)))  unsigned int u32x2;
typedef __attribute__((ext_vector_type(8)))  int          i32x8;
typedef __attribute__((ext_vector_type(4)))  int          i32x4;

#define B_DIM 8192
#define D_DIM 1024
#define BT    128     // block tile (M and N)
#define KT    32      // K step (bf16 WMMA depth)
#define LDSTR 40      // padded LDS row stride in ushorts (80B: conflict-free, 16B aligned)
#define TILE_USH   (BT * LDSTR)          // 5120 ushorts per tile
#define BUF_USH    (4 * TILE_USH)        // Ahi,Alo,Bhi,Blo per buffer
#define TILE_BYTES (TILE_USH * 2)        // 10240
#define BUF_BYTES  (BUF_USH * 2)         // 40960

union Frag { v16bf v; u32x4 q[2]; };

// monotone float<->uint encoding so unsigned atomicMin == float min
__device__ __forceinline__ unsigned f2o(float f) {
  unsigned u = __float_as_uint(f);
  return (u & 0x80000000u) ? ~u : (u | 0x80000000u);
}
__device__ __forceinline__ float o2f(unsigned u) {
  return (u & 0x80000000u) ? __uint_as_float(u & 0x7FFFFFFFu) : __uint_as_float(~u);
}

// round-to-nearest-even split of f32 into two bf16 halves (x ~= hi + lo)
__device__ __forceinline__ void bf16_split(float x, unsigned short& h, unsigned short& l) {
  unsigned u  = __float_as_uint(x);
  unsigned hu = (u + 0x7FFFu + ((u >> 16) & 1u)) >> 16;
  float    r  = x - __uint_as_float(hu << 16);
  unsigned v  = __float_as_uint(r);
  h = (unsigned short)hu;
  l = (unsigned short)((v + 0x7FFFu + ((v >> 16) & 1u)) >> 16);
}

__device__ __forceinline__ u32x2 pack4(const unsigned short s[4]) {
  u32x2 p;
  p.x = (unsigned)s[0] | ((unsigned)s[1] << 16);
  p.y = (unsigned)s[2] | ((unsigned)s[3] << 16);
  return p;
}

__device__ __forceinline__ v8f wmma_bf16(v16bf a, v16bf b, v8f c) {
  return __builtin_amdgcn_wmma_f32_16x16x32_bf16(false, a, false, b, (short)0, c, false, false);
}

// ---- shared epilogue: acc tiles -> block row/col minima -> global ---------
__device__ __forceinline__ void min_epilogue(
    v8f acc[2][4], unsigned short* smem_u16,
    const float* __restrict__ sqx, const float* __restrict__ sqy,
    unsigned* __restrict__ rming, unsigned* __restrict__ cming,
    int blockM, int blockN, int rowBase, int colBase, int half, int l16, int tid) {
  unsigned* lmin = (unsigned*)smem_u16;  // [0..127]=rows, [128..255]=cols
  lmin[tid] = 0xFFFFFFFFu;
  __syncthreads();

  const float inv_d = 1.0f / (float)D_DIM;
  float sy[4];
#pragma unroll
  for (int ni = 0; ni < 4; ++ni) sy[ni] = sqy[blockN + colBase + ni * 16 + l16];
  float colmin_l[4];
#pragma unroll
  for (int ni = 0; ni < 4; ++ni) colmin_l[ni] = 3.0e38f;

#pragma unroll
  for (int mi = 0; mi < 2; ++mi) {
#pragma unroll
    for (int r = 0; r < 8; ++r) {
      const int mrow = rowBase + mi * 16 + half * 8 + r;
      const float sx = sqx[blockM + mrow];
      float rm = 3.0e38f;
#pragma unroll
      for (int ni = 0; ni < 4; ++ni) {
        float mse = (sx + sy[ni] - 2.0f * acc[mi][ni][r]) * inv_d;
        rm = fminf(rm, mse);
        colmin_l[ni] = fminf(colmin_l[ni], mse);
      }
#pragma unroll
      for (int m = 1; m <= 8; m <<= 1) rm = fminf(rm, __shfl_xor(rm, m, 32));
      if (l16 == 0) atomicMin(&lmin[mrow], f2o(rm));
    }
  }
#pragma unroll
  for (int ni = 0; ni < 4; ++ni) {
    float cm = fminf(colmin_l[ni], __shfl_xor(colmin_l[ni], 16, 32));
    if (half == 0) atomicMin(&lmin[128 + colBase + ni * 16 + l16], f2o(cm));
  }
  __syncthreads();
  if (tid < 128) atomicMin(&rming[blockM + tid], lmin[tid]);
  else           atomicMin(&cming[blockN + tid - 128], lmin[tid]);
}

// ---- kernel 1: row squared norms + init min buffers -----------------------
__global__ __launch_bounds__(256)
void sqnorm_init_kernel(const float* __restrict__ X, const float* __restrict__ Y,
                        float* __restrict__ sqx, float* __restrict__ sqy,
                        unsigned* __restrict__ rming, unsigned* __restrict__ cming) {
  const int lane = threadIdx.x & 31;
  const int wave = threadIdx.x >> 5;
  const int row  = blockIdx.x * 8 + wave;  // 0..16383 over 2048 blocks
  const float* src; float* dst; unsigned* mp;
  if (row < B_DIM) { src = X + (size_t)row * D_DIM; dst = sqx + row; mp = rming + row; }
  else { int r2 = row - B_DIM; src = Y + (size_t)r2 * D_DIM; dst = sqy + r2; mp = cming + r2; }
  float s = 0.f;
#pragma unroll
  for (int i = 0; i < 8; ++i) {
    f32x4 v = *(const f32x4*)(src + (size_t)(i * 32 + lane) * 4);
    s += v.x * v.x + v.y * v.y + v.z * v.z + v.w * v.w;
  }
#pragma unroll
  for (int m = 16; m >= 1; m >>= 1) s += __shfl_xor(s, m, 32);
  if (lane == 0) { *dst = s; *mp = 0xFFFFFFFFu; }
}

// ---- kernel 2a (fallback): convert-in-kernel bf16x3 WMMA GEMM -------------
__global__ __launch_bounds__(256)
void pairwise_min_gemm(const float* __restrict__ X, const float* __restrict__ Y,
                       const float* __restrict__ sqx, const float* __restrict__ sqy,
                       unsigned* __restrict__ rming, unsigned* __restrict__ cming) {
  __shared__ __align__(16) unsigned short smem[4 * BT * LDSTR];  // 40 KB
  unsigned short* Ahi = smem;
  unsigned short* Alo = smem + 1 * TILE_USH;
  unsigned short* Bhi = smem + 2 * TILE_USH;
  unsigned short* Blo = smem + 3 * TILE_USH;

  const int tid = threadIdx.x, lane = tid & 31, wave = tid >> 5;
  const int waveM = wave & 3, waveN = wave >> 2;
  const int blockM = blockIdx.x * BT, blockN = blockIdx.y * BT;
  const int rowBase = waveM * 32, colBase = waveN * 64;
  const int half = lane >> 4, l16 = lane & 15;

  v8f zero = {};
  v8f acc[2][4];
#pragma unroll
  for (int mi = 0; mi < 2; ++mi)
#pragma unroll
    for (int ni = 0; ni < 4; ++ni) acc[mi][ni] = zero;

  const int ldRow = tid >> 3;
  const int ldK   = (tid & 7) << 2;

  for (int k0 = 0; k0 < D_DIM; k0 += KT) {
#pragma unroll
    for (int i = 0; i < 4; ++i) {
      const int r = ldRow + i * 32;
      f32x4 va = *(const f32x4*)(X + (size_t)(blockM + r) * D_DIM + k0 + ldK);
      f32x4 vb = *(const f32x4*)(Y + (size_t)(blockN + r) * D_DIM + k0 + ldK);
      unsigned short ah[4], al[4], bh[4], bl[4];
      bf16_split(va.x, ah[0], al[0]); bf16_split(va.y, ah[1], al[1]);
      bf16_split(va.z, ah[2], al[2]); bf16_split(va.w, ah[3], al[3]);
      bf16_split(vb.x, bh[0], bl[0]); bf16_split(vb.y, bh[1], bl[1]);
      bf16_split(vb.z, bh[2], bl[2]); bf16_split(vb.w, bh[3], bl[3]);
      *(u32x2*)(Ahi + r * LDSTR + ldK) = pack4(ah);
      *(u32x2*)(Alo + r * LDSTR + ldK) = pack4(al);
      *(u32x2*)(Bhi + r * LDSTR + ldK) = pack4(bh);
      *(u32x2*)(Blo + r * LDSTR + ldK) = pack4(bl);
    }
    __syncthreads();

    Frag fah[2], fal[2], fbh[4], fbl[4];
#pragma unroll
    for (int mi = 0; mi < 2; ++mi) {
      const unsigned short* pa = Ahi + (rowBase + mi * 16 + l16) * LDSTR + (half << 3);
      fah[mi].q[0] = *(const u32x4*)pa;  fah[mi].q[1] = *(const u32x4*)(pa + 16);
      const unsigned short* pl = Alo + (rowBase + mi * 16 + l16) * LDSTR + (half << 3);
      fal[mi].q[0] = *(const u32x4*)pl;  fal[mi].q[1] = *(const u32x4*)(pl + 16);
    }
#pragma unroll
    for (int ni = 0; ni < 4; ++ni) {
      const unsigned short* pb = Bhi + (colBase + ni * 16 + l16) * LDSTR + (half << 3);
      fbh[ni].q[0] = *(const u32x4*)pb;  fbh[ni].q[1] = *(const u32x4*)(pb + 16);
      const unsigned short* pl = Blo + (colBase + ni * 16 + l16) * LDSTR + (half << 3);
      fbl[ni].q[0] = *(const u32x4*)pl;  fbl[ni].q[1] = *(const u32x4*)(pl + 16);
    }
#pragma unroll
    for (int mi = 0; mi < 2; ++mi)
#pragma unroll
      for (int ni = 0; ni < 4; ++ni) {
        acc[mi][ni] = wmma_bf16(fah[mi].v, fbh[ni].v, acc[mi][ni]);
        acc[mi][ni] = wmma_bf16(fah[mi].v, fbl[ni].v, acc[mi][ni]);
        acc[mi][ni] = wmma_bf16(fal[mi].v, fbh[ni].v, acc[mi][ni]);
      }
    __syncthreads();
  }
  min_epilogue(acc, smem, sqx, sqy, rming, cming, blockM, blockN, rowBase, colBase, half, l16, tid);
}

#if __has_builtin(__builtin_amdgcn_tensor_load_to_lds) && __has_builtin(__builtin_amdgcn_s_wait_tensorcnt)
#define HAVE_TDM 1

// ---- kernel P: one-time f32 -> (hi,lo) bf16 conversion into workspace -----
__global__ __launch_bounds__(256)
void convert_kernel(const float* __restrict__ X, const float* __restrict__ Y,
                    unsigned short* __restrict__ Ahi, unsigned short* __restrict__ Alo,
                    unsigned short* __restrict__ Bhi, unsigned short* __restrict__ Blo) {
  const size_t i4 = (size_t)blockIdx.x * 256 + threadIdx.x;  // one f32x4 of X and of Y
  f32x4 a = *(const f32x4*)(X + i4 * 4);
  f32x4 b = *(const f32x4*)(Y + i4 * 4);
  unsigned short ah[4], al[4], bh[4], bl[4];
  bf16_split(a.x, ah[0], al[0]); bf16_split(a.y, ah[1], al[1]);
  bf16_split(a.z, ah[2], al[2]); bf16_split(a.w, ah[3], al[3]);
  bf16_split(b.x, bh[0], bl[0]); bf16_split(b.y, bh[1], bl[1]);
  bf16_split(b.z, bh[2], bl[2]); bf16_split(b.w, bh[3], bl[3]);
  *(u32x2*)(Ahi + i4 * 4) = pack4(ah);
  *(u32x2*)(Alo + i4 * 4) = pack4(al);
  *(u32x2*)(Bhi + i4 * 4) = pack4(bh);
  *(u32x2*)(Blo + i4 * 4) = pack4(bl);
}

// TDM 2D tile load: BT x KT bf16 tile -> LDS, with 16B pad after each 64B row
// so the LDS row stride is LDSTR ushorts (bank-conflict-free fragment reads).
__device__ __forceinline__ void tdm_load_tile(unsigned lds_byte_off,
                                              const unsigned short* gsrc) {
  unsigned long long ga = (unsigned long long)(uintptr_t)gsrc;
  u32x4 g0;
  g0.x = 1u;                                                 // count=1 user descriptor
  g0.y = lds_byte_off;                                       // lds_addr (bytes)
  g0.z = (unsigned)(ga & 0xFFFFFFFFu);                       // global_addr[31:0]
  g0.w = (unsigned)((ga >> 32) & 0x1FFFFFFu) | 0x80000000u;  // addr[56:32] | type=2
  i32x8 g1;
  g1[0] = (int)((1u << 16)      // data_size = 2 bytes
              | (1u << 20)      // pad_enable
              | (3u << 22)      // pad_interval: 16 DWORDs (64B) between pads
              | (3u << 25));    // pad_amount: 4 DWORDs (16B)
  g1[1] = (int)(((unsigned)D_DIM & 0xFFFFu) << 16);          // tensor_dim0 lo16
  g1[2] = (int)(((unsigned)D_DIM >> 16) | (((unsigned)B_DIM & 0xFFFFu) << 16)); // dim0 hi / dim1 lo
  g1[3] = (int)(((unsigned)B_DIM >> 16) | ((unsigned)KT << 16));                // dim1 hi / tile_dim0
  g1[4] = (int)BT;                                           // tile_dim1=128, tile_dim2=0
  g1[5] = (int)D_DIM;                                        // tensor_dim0_stride lo32
  g1[6] = 0;
  g1[7] = 0;
  i32x4 z4 = {0, 0, 0, 0};
#if __clang_major__ >= 23
  i32x8 z8 = {0, 0, 0, 0, 0, 0, 0, 0};
  __builtin_amdgcn_tensor_load_to_lds(g0, g1, z4, z4, z8, 0);
#else
  __builtin_amdgcn_tensor_load_to_lds(g0, g1, z4, z4, 0);
#endif
}

// ---- kernel 2b: TDM double-buffered bf16x3 WMMA GEMM ----------------------
__global__ __launch_bounds__(256)
void pairwise_min_gemm_tdm(const unsigned short* __restrict__ Ahi, const unsigned short* __restrict__ Alo,
                           const unsigned short* __restrict__ Bhi, const unsigned short* __restrict__ Blo,
                           const float* __restrict__ sqx, const float* __restrict__ sqy,
                           unsigned* __restrict__ rming, unsigned* __restrict__ cming) {
  extern __shared__ __align__(16) unsigned short dynsmem[];  // 2 * BUF_BYTES = 80 KB

  const int tid = threadIdx.x, lane = tid & 31, wave = tid >> 5;
  const int waveM = wave & 3, waveN = wave >> 2;
  const int blockM = blockIdx.x * BT, blockN = blockIdx.y * BT;
  const int rowBase = waveM * 32, colBase = waveN * 64;
  const int half = lane >> 4, l16 = lane & 15;

  // waves 0..3 own one TDM tile each: 0=Ahi 1=Alo 2=Bhi 3=Blo
  const unsigned short* gsrc0 = nullptr;
  if (wave < 4) {
    const unsigned short* mats[4] = {Ahi, Alo, Bhi, Blo};
    const int rb = (wave < 2) ? blockM : blockN;
    gsrc0 = mats[wave] + (size_t)rb * D_DIM;
  }

  v8f zero = {};
  v8f acc[2][4];
#pragma unroll
  for (int mi = 0; mi < 2; ++mi)
#pragma unroll
    for (int ni = 0; ni < 4; ++ni) acc[mi][ni] = zero;

  const int NSTEP = D_DIM / KT;
  if (wave < 4) tdm_load_tile((unsigned)(wave * TILE_BYTES), gsrc0);  // k=0 -> buf0

  int cur = 0;
  for (int s = 0; s < NSTEP; ++s) {
    if (wave < 4) {
      if (s + 1 < NSTEP) {
        tdm_load_tile((unsigned)((cur ^ 1) * BUF_BYTES + wave * TILE_BYTES),
                      gsrc0 + (s + 1) * KT);
        __builtin_amdgcn_s_wait_tensorcnt(1);  // current buffer done; prefetch in flight
      } else {
        __builtin_amdgcn_s_wait_tensorcnt(0);  // last step: drain
      }
    }
    __syncthreads();  // buf[cur] ready for everyone; prior reads of buf[cur^1] done

    const unsigned short* base = dynsmem + cur * BUF_USH;
    const unsigned short* tAhi = base;
    const unsigned short* tAlo = base + 1 * TILE_USH;
    const unsigned short* tBhi = base + 2 * TILE_USH;
    const unsigned short* tBlo = base + 3 * TILE_USH;

    Frag fah[2], fal[2], fbh[4], fbl[4];
#pragma unroll
    for (int mi = 0; mi < 2; ++mi) {
      const unsigned short* pa = tAhi + (rowBase + mi * 16 + l16) * LDSTR + (half << 3);
      fah[mi].q[0] = *(const u32x4*)pa;  fah[mi].q[1] = *(const u32x4*)(pa + 16);
      const unsigned short* pl = tAlo + (rowBase + mi * 16 + l16) * LDSTR + (half << 3);
      fal[mi].q[0] = *(const u32x4*)pl;  fal[mi].q[1] = *(const u32x4*)(pl + 16);
    }
#pragma unroll
    for (int ni = 0; ni < 4; ++ni) {
      const unsigned short* pb = tBhi + (colBase + ni * 16 + l16) * LDSTR + (half << 3);
      fbh[ni].q[0] = *(const u32x4*)pb;  fbh[ni].q[1] = *(const u32x4*)(pb + 16);
      const unsigned short* pl = tBlo + (colBase + ni * 16 + l16) * LDSTR + (half << 3);
      fbl[ni].q[0] = *(const u32x4*)pl;  fbl[ni].q[1] = *(const u32x4*)(pl + 16);
    }
#pragma unroll
    for (int mi = 0; mi < 2; ++mi)
#pragma unroll
      for (int ni = 0; ni < 4; ++ni) {
        acc[mi][ni] = wmma_bf16(fah[mi].v, fbh[ni].v, acc[mi][ni]);
        acc[mi][ni] = wmma_bf16(fah[mi].v, fbl[ni].v, acc[mi][ni]);
        acc[mi][ni] = wmma_bf16(fal[mi].v, fbh[ni].v, acc[mi][ni]);
      }
    __syncthreads();  // all reads of buf[cur] done before it is refilled
    cur ^= 1;
  }
  min_epilogue(acc, dynsmem, sqx, sqy, rming, cming, blockM, blockN, rowBase, colBase, half, l16, tid);
}
#endif  // HAVE_TDM

// ---- kernel 3: mean of the 16384 minima -----------------------------------
__global__ __launch_bounds__(256)
void finalize_kernel(const unsigned* __restrict__ rming, const unsigned* __restrict__ cming,
                     float* __restrict__ out) {
  __shared__ float red[256];
  float s = 0.f;
  for (int i = threadIdx.x; i < B_DIM; i += 256) s += o2f(rming[i]) + o2f(cming[i]);
  red[threadIdx.x] = s;
  __syncthreads();
  for (int off = 128; off > 0; off >>= 1) {
    if (threadIdx.x < off) red[threadIdx.x] += red[threadIdx.x + off];
    __syncthreads();
  }
  if (threadIdx.x == 0) out[0] = red[0] / (float)(2 * B_DIM);
}

// ---- launcher --------------------------------------------------------------
extern "C" void kernel_launch(void* const* d_in, const int* in_sizes, int n_in,
                              void* d_out, int out_size, void* d_ws, size_t ws_size,
                              hipStream_t stream) {
  const float* X = (const float*)d_in[0];
  const float* Y = (const float*)d_in[1];
  float*    sqx  = (float*)d_ws;
  float*    sqy  = sqx + B_DIM;
  unsigned* rmin = (unsigned*)(sqy + B_DIM);
  unsigned* cmin = rmin + B_DIM;
  float*    out  = (float*)d_out;

  const size_t head_bytes = 4ull * B_DIM * 4ull;                 // 128 KB
  const size_t mat_elems  = (size_t)B_DIM * D_DIM;
  const size_t need_tdm   = head_bytes + 4ull * mat_elems * 2ull;  // + 64 MB

  sqnorm_init_kernel<<<2 * B_DIM / 8, 256, 0, stream>>>(X, Y, sqx, sqy, rmin, cmin);

#if defined(HAVE_TDM)
  if (ws_size >= need_tdm) {
    unsigned short* Ahi = (unsigned short*)((char*)d_ws + head_bytes);
    unsigned short* Alo = Ahi + mat_elems;
    unsigned short* Bhi = Alo + mat_elems;
    unsigned short* Blo = Bhi + mat_elems;
    convert_kernel<<<(unsigned)(mat_elems / 4 / 256), 256, 0, stream>>>(X, Y, Ahi, Alo, Bhi, Blo);
    pairwise_min_gemm_tdm<<<dim3(B_DIM / BT, B_DIM / BT), 256, 2 * BUF_BYTES, stream>>>(
        Ahi, Alo, Bhi, Blo, sqx, sqy, rmin, cmin);
  } else
#endif
  {
    pairwise_min_gemm<<<dim3(B_DIM / BT, B_DIM / BT), 256, 0, stream>>>(X, Y, sqx, sqy, rmin, cmin);
  }
  finalize_kernel<<<1, 256, 0, stream>>>(rmin, cmin, out);
}
